// DyMEAN_77266461655555
// MI455X (gfx1250) — matrix operand order repaired and checked by reading, not compiled, not run on hardware
//
#include <hip/hip_runtime.h>
#include <hip/hip_bf16.h>
#include <stdint.h>

// ---------------------------------------------------------------------------
// DyMEAN-style EGNN, fused for MI455X (gfx1250): wave32 + WMMA bf16 16x16x32.
// Edge tiles of 16 edges -> entire edge MLP (544->256->256->256->16) runs as
// WMMA chains with LDS-staged bf16 activations; scatter via global atomics.
// ---------------------------------------------------------------------------

typedef __attribute__((ext_vector_type(16))) __bf16 v16bf;
typedef __attribute__((ext_vector_type(8)))  float  v8f;

#define NCH   4
#define CFX   16
#define DF    256
#define HIDX  256
#define EDIM  16
#define CIN1  544   // 2*DF + CFX + EDIM
#define CINN  512   // DF + HIDX

__device__ __forceinline__ unsigned short f2bf(float f) {
  unsigned int u = __float_as_uint(f);
  u += 0x7FFFu + ((u >> 16) & 1u);          // round-to-nearest-even
  return (unsigned short)(u >> 16);
}
__device__ __forceinline__ float siluf(float v) { return v / (1.0f + __expf(-v)); }

union FragU { v16bf v; uint4 q[2]; };

// A-matrix 16x32 bf16 fragment from LDS row-major tile.
// lanes 0-15: M=lane, K pairs {0..7, 16..23}; lanes 16-31: M=lane-16, {8..15, 24..31}
__device__ __forceinline__ v16bf frag_a_lds(const unsigned short* base, int strideElems,
                                            int kBase, int lane) {
  const int m   = lane & 15;
  const int sel = (lane >> 4) & 1;
  const unsigned short* p = base + m * strideElems + kBase + sel * 8;
  FragU r;
  r.q[0] = *(const uint4*)(p);
  r.q[1] = *(const uint4*)(p + 16);
  return r.v;
}

// B-matrix 32x16 bf16 fragment from pre-transposed weights Wt[N][Ktot] in global.
// lanes 0-15: N=lane, K {0..7, 16..23}; lanes 16-31: N=lane-16, K {8..15, 24..31}
__device__ __forceinline__ v16bf frag_b_glb(const unsigned short* wT, int Ktot,
                                            int nBase, int kBase, int lane) {
  const int n   = (lane & 15) + nBase;
  const int sel = (lane >> 4) & 1;
  const unsigned short* p = wT + (size_t)n * Ktot + kBase + sel * 8;
  FragU r;
  r.q[0] = *(const uint4*)(p);
  r.q[1] = *(const uint4*)(p + 16);
  return r.v;
}

template <int KTOT>
__device__ __forceinline__ v8f wave_gemm(const unsigned short* aLds, int aStride,
                                         const unsigned short* wT, int nBase, int lane) {
  v8f acc = {0.f, 0.f, 0.f, 0.f, 0.f, 0.f, 0.f, 0.f};
#pragma unroll
  for (int ks = 0; ks < KTOT; ks += 32) {
    v16bf a = frag_a_lds(aLds, aStride, ks, lane);
    v16bf b = frag_b_glb(wT, KTOT, nBase, ks, lane);
    acc = __builtin_amdgcn_wmma_f32_16x16x32_bf16(false, a, false, b,
                                                  (short)0, acc, false, false);
  }
  return acc;
}

// ---------------------------------------------------------------------------
// Edge kernel: one 16-edge tile per 128-thread block (4 wave32s).
// ---------------------------------------------------------------------------
__global__ __launch_bounds__(128, 1)
void dymean_edge_kernel(const float* __restrict__ h, const float* __restrict__ x,
                        const long long* __restrict__ edge_list,
                        const float* __restrict__ channel_attr,
                        const float* __restrict__ channel_weights,
                        const float* __restrict__ edge_weights,
                        const float* __restrict__ edge_attr,
                        const unsigned short* __restrict__ w_radT, const float* __restrict__ b_rad,
                        const unsigned short* __restrict__ w_e1T,  const float* __restrict__ b_e1,
                        const unsigned short* __restrict__ w_e2T,  const float* __restrict__ b_e2,
                        const unsigned short* __restrict__ w_c1T,  const float* __restrict__ b_c1,
                        const unsigned short* __restrict__ w_c2T,
                        float* __restrict__ agg, float* __restrict__ xacc, int nEdges) {
  __shared__ int   s_row[16], s_col[16], s_valid[16];
  __shared__ float s_ew[16];
  __shared__ float s_cd[16][NCH][NCH][3];
  __shared__ float s_rad[16][NCH][NCH];
  __shared__ float s_ar[16][NCH][CFX];
  __shared__ float s_ac[16][NCH][CFX];
  __shared__ float s_t[16][NCH][CFX];
  __shared__ __align__(16) unsigned short s_A1[16][CIN1];
  __shared__ __align__(16) unsigned short s_A2[16][HIDX];
  __shared__ __align__(16) unsigned short s_A3[16][HIDX];
  __shared__ float s_phi[16][16];

  const int tid  = threadIdx.x;
  const int lane = tid & 31;
  const int wave = tid >> 5;
  const int tileBase = blockIdx.x * 16;

  if (tid < 16) {
    int e  = tileBase + tid;
    int ok = (e < nEdges) ? 1 : 0;
    int ec = ok ? e : 0;
    s_row[tid]   = (int)edge_list[(size_t)ec * 3 + 0];
    s_col[tid]   = (int)edge_list[(size_t)ec * 3 + 1];
    s_ew[tid]    = ok ? edge_weights[ec] : 0.f;
    s_valid[tid] = ok;
  }
  __syncthreads();

  // geometry: coordinate deltas + masked radial distances
  for (int i = tid; i < 256; i += 128) {
    int e = i >> 4, p = i & 15, c = p >> 2, d = p & 3;
    int rn = s_row[e], cn = s_col[e];
    float r2 = 0.f;
#pragma unroll
    for (int k = 0; k < 3; ++k) {
      float dv = x[((size_t)rn * NCH + c) * 3 + k] - x[((size_t)cn * NCH + d) * 3 + k];
      s_cd[e][c][d][k] = dv;
      r2 += dv * dv;
    }
    float cm = channel_weights[(size_t)rn * NCH + c] * channel_weights[(size_t)cn * NCH + d];
    s_rad[e][c][d] = r2 * cm;
  }
  for (int i = tid; i < 16 * NCH * CFX; i += 128) {
    int e = i >> 6, q = i & 63, c = q >> 4, a = q & 15;
    s_ar[e][c][a] = channel_attr[((size_t)s_row[e] * NCH + c) * CFX + a];
    s_ac[e][c][a] = channel_attr[((size_t)s_col[e] * NCH + c) * CFX + a];
  }
  __syncthreads();

  // t[e][c][b] = sum_d radial[c][d] * attr_col[d][b]
  for (int i = tid; i < 16 * NCH * CFX; i += 128) {
    int e = i >> 6, q = i & 63, c = q >> 4, b = q & 15;
    float acc = 0.f;
#pragma unroll
    for (int d = 0; d < NCH; ++d) acc += s_rad[e][c][d] * s_ac[e][d][b];
    s_t[e][c][b] = acc;
  }
  __syncthreads();

  // R[e][a*16+b] = sum_c attr_row[c][a] * t[c][b]  -> bf16 A-tile (16x256)
  for (int i = tid; i < 16 * 256; i += 128) {
    int e = i >> 8, q = i & 255, a = q >> 4, b = q & 15;
    float acc = 0.f;
#pragma unroll
    for (int c = 0; c < NCH; ++c) acc += s_ar[e][c][a] * s_t[e][c][b];
    s_A2[e][q] = f2bf(acc);
  }
  __syncthreads();

  // wave0: rfeat = R @ w_rad + b_rad (WMMA, N=16);  waves1-3: gather h/edge_attr
  if (wave == 0) {
    v8f acc = wave_gemm<256>(&s_A2[0][0], HIDX, w_radT, 0, lane);
    int n = lane & 15, mb = (lane >> 4) * 8;
    float bias = b_rad[n];
#pragma unroll
    for (int r = 0; r < 8; ++r) s_A1[mb + r][512 + n] = f2bf(acc[r] + bias);
  } else {
    for (int i = tid - 32; i < 16 * 528; i += 96) {
      int e = i / 528, j = i - e * 528;
      float v;
      if (j < 256)      v = h[(size_t)s_row[e] * DF + j];
      else if (j < 512) v = h[(size_t)s_col[e] * DF + (j - 256)];
      else              v = s_valid[e] ? edge_attr[(size_t)(tileBase + e) * EDIM + (j - 512)] : 0.f;
      int col = (j < 512) ? j : (j + 16);   // skip the 16 rfeat columns
      s_A1[e][col] = f2bf(v);
    }
  }
  __syncthreads();

  // GEMM1: (16x544) @ w_e1 -> silu -> s_A2
  for (int nt = wave * 4; nt < wave * 4 + 4; ++nt) {
    v8f acc = wave_gemm<CIN1>(&s_A1[0][0], CIN1, w_e1T, nt * 16, lane);
    int n = lane & 15, mb = (lane >> 4) * 8;
    float bias = b_e1[nt * 16 + n];
#pragma unroll
    for (int r = 0; r < 8; ++r) s_A2[mb + r][nt * 16 + n] = f2bf(siluf(acc[r] + bias));
  }
  __syncthreads();

  // GEMM2: @ w_e2 -> silu -> * edge_weight -> m ; atomic scatter into agg[row]
  for (int nt = wave * 4; nt < wave * 4 + 4; ++nt) {
    v8f acc = wave_gemm<HIDX>(&s_A2[0][0], HIDX, w_e2T, nt * 16, lane);
    int n = lane & 15, mb = (lane >> 4) * 8;
    float bias = b_e2[nt * 16 + n];
#pragma unroll
    for (int r = 0; r < 8; ++r) {
      int m = mb + r;
      float v = siluf(acc[r] + bias) * s_ew[m];
      s_A3[m][nt * 16 + n] = f2bf(v);
      if (s_valid[m]) atomicAdd(&agg[(size_t)s_row[m] * HIDX + nt * 16 + n], v);
    }
  }
  __syncthreads();

  // GEMM c1: @ w_c1 -> silu -> s_A2
  for (int nt = wave * 4; nt < wave * 4 + 4; ++nt) {
    v8f acc = wave_gemm<HIDX>(&s_A3[0][0], HIDX, w_c1T, nt * 16, lane);
    int n = lane & 15, mb = (lane >> 4) * 8;
    float bias = b_c1[nt * 16 + n];
#pragma unroll
    for (int r = 0; r < 8; ++r) s_A2[mb + r][nt * 16 + n] = f2bf(siluf(acc[r] + bias));
  }
  __syncthreads();

  // GEMM c2: @ w_c2 -> phi (16x16, no bias), wave0 only
  if (wave == 0) {
    v8f acc = wave_gemm<HIDX>(&s_A2[0][0], HIDX, w_c2T, 0, lane);
    int n = lane & 15, mb = (lane >> 4) * 8;
#pragma unroll
    for (int r = 0; r < 8; ++r) s_phi[mb + r][n] = acc[r];
  }
  __syncthreads();

  // trans[e][c][k] = sum_d cd[e][c][d][k]*phi[e][c*4+d] -> atomic scatter into xacc[row]
  for (int i = tid; i < 16 * NCH * 3; i += 128) {
    int e = i / 12, rem = i - e * 12, c = rem / 3, k = rem - c * 3;
    if (!s_valid[e]) continue;
    float tr = 0.f;
#pragma unroll
    for (int d = 0; d < NCH; ++d) tr += s_cd[e][c][d][k] * s_phi[e][c * NCH + d];
    atomicAdd(&xacc[((size_t)s_row[e] * NCH + c) * 3 + k], tr);
  }
}

// ---------------------------------------------------------------------------
// Node kernel: one 16-node tile per 128-thread block. nh MLP + residual + LN,
// plus x update with degree normalization.
// ---------------------------------------------------------------------------
__global__ __launch_bounds__(128, 1)
void dymean_node_kernel(const float* __restrict__ h_in, const float* __restrict__ x_in,
                        const float* __restrict__ agg, const float* __restrict__ xacc,
                        const float* __restrict__ deg,
                        const unsigned short* __restrict__ w_n1T, const float* __restrict__ b_n1,
                        const unsigned short* __restrict__ w_n2T, const float* __restrict__ b_n2,
                        const float* __restrict__ ln_g, const float* __restrict__ ln_b,
                        float* __restrict__ h_out, float* __restrict__ x_out, int nNodes) {
  __shared__ __align__(16) unsigned short s_A[16][CINN];
  __shared__ __align__(16) unsigned short s_H[16][HIDX];
  __shared__ float s_nh[16][DF];
  __shared__ float s_sum[16][8], s_sq[16][8];
  __shared__ float s_mu[16], s_ri[16];

  const int tid = threadIdx.x, lane = tid & 31, wave = tid >> 5;
  const int base = blockIdx.x * 16;

  for (int i = tid; i < 16 * CINN; i += 128) {
    int e = i >> 9, j = i & 511;
    int node = base + e; if (node >= nNodes) node = nNodes - 1;
    float v = (j < DF) ? h_in[(size_t)node * DF + j] : agg[(size_t)node * HIDX + (j - DF)];
    s_A[e][j] = f2bf(v);
  }
  __syncthreads();

  for (int nt = wave * 4; nt < wave * 4 + 4; ++nt) {
    v8f acc = wave_gemm<CINN>(&s_A[0][0], CINN, w_n1T, nt * 16, lane);
    int n = lane & 15, mb = (lane >> 4) * 8;
    float bias = b_n1[nt * 16 + n];
#pragma unroll
    for (int r = 0; r < 8; ++r) s_H[mb + r][nt * 16 + n] = f2bf(siluf(acc[r] + bias));
  }
  __syncthreads();

  for (int nt = wave * 4; nt < wave * 4 + 4; ++nt) {
    v8f acc = wave_gemm<HIDX>(&s_H[0][0], HIDX, w_n2T, nt * 16, lane);
    int n = lane & 15, mb = (lane >> 4) * 8;
    float bias = b_n2[nt * 16 + n];
#pragma unroll
    for (int r = 0; r < 8; ++r) {
      int node = base + mb + r; if (node >= nNodes) node = nNodes - 1;
      s_nh[mb + r][nt * 16 + n] = acc[r] + bias + h_in[(size_t)node * DF + nt * 16 + n];
    }
  }
  __syncthreads();

  { // layernorm partial sums: 8 threads per node
    int e = tid >> 3, s = tid & 7;
    float sm = 0.f, sq = 0.f;
    for (int f = s * 32; f < s * 32 + 32; ++f) { float v = s_nh[e][f]; sm += v; sq += v * v; }
    s_sum[e][s] = sm; s_sq[e][s] = sq;
  }
  __syncthreads();
  if (tid < 16) {
    float sm = 0.f, sq = 0.f;
    for (int s = 0; s < 8; ++s) { sm += s_sum[tid][s]; sq += s_sq[tid][s]; }
    float mu = sm / (float)DF;
    float var = sq / (float)DF - mu * mu;
    s_mu[tid] = mu;
    s_ri[tid] = rsqrtf(var + 1e-5f);
  }
  __syncthreads();
  for (int i = tid; i < 16 * DF; i += 128) {
    int e = i >> 8, f = i & 255;
    int node = base + e;
    if (node < nNodes)
      h_out[(size_t)node * DF + f] = (s_nh[e][f] - s_mu[e]) * s_ri[e] * ln_g[f] + ln_b[f];
  }
  for (int i = tid; i < 16 * 12; i += 128) {
    int e = i / 12, rem = i - e * 12;
    int node = base + e;
    if (node < nNodes) {
      float dg = deg[node]; if (dg < 1.f) dg = 1.f;
      size_t idx = (size_t)node * 12 + rem;
      x_out[idx] = x_in[idx] + xacc[idx] / dg;
    }
  }
}

// ---------------------------------------------------------------------------
// utility kernels
// ---------------------------------------------------------------------------
__global__ void zero_f32(float* p, size_t n) {
  size_t i = (size_t)blockIdx.x * blockDim.x + threadIdx.x;
  size_t stride = (size_t)gridDim.x * blockDim.x;
  for (; i < n; i += stride) p[i] = 0.f;
}
__global__ void degree_kernel(const long long* __restrict__ el, float* deg, int nEdges) {
  int i = blockIdx.x * blockDim.x + threadIdx.x;
  if (i < nEdges) atomicAdd(&deg[(int)el[(size_t)i * 3]], 1.0f);
}
// W[K][N] f32 -> Wt[N][K] bf16 (so B fragments are K-contiguous per output col)
__global__ void transpose_bf16(const float* __restrict__ W, unsigned short* __restrict__ Wt,
                               int K, int Nn) {
  int i = blockIdx.x * blockDim.x + threadIdx.x;
  if (i < K * Nn) {
    int k = i / Nn, n = i - k * Nn;
    Wt[(size_t)n * K + k] = f2bf(W[i]);
  }
}

// ---------------------------------------------------------------------------
// d_in layout: 0 input, 1 edge_list(i64), 2 coords, 3 channel_attr,
// 4 channel_weights, 5 edge_weights, 6 edge_attr, then params flattened in
// sorted key order per layer (15 tensors x 3 layers):
// b_c1,b_e1,b_e2,b_n1,b_n2,b_rad,ln_b,ln_g,w_c1,w_c2,w_e1,w_e2,w_n1,w_n2,w_rad
// ---------------------------------------------------------------------------
extern "C" void kernel_launch(void* const* d_in, const int* in_sizes, int n_in,
                              void* d_out, int out_size, void* d_ws, size_t ws_size,
                              hipStream_t stream) {
  (void)n_in; (void)out_size; (void)ws_size;
  const float*     h0    = (const float*)d_in[0];
  const long long* el    = (const long long*)d_in[1];
  const float*     x0    = (const float*)d_in[2];
  const float*     cattr = (const float*)d_in[3];
  const float*     cw    = (const float*)d_in[4];
  const float*     ew    = (const float*)d_in[5];
  const float*     eattr = (const float*)d_in[6];
  const int N = in_sizes[0] / DF;
  const int E = in_sizes[5];

  char* ws = (char*)d_ws;
  size_t off = 0;
  auto carve = [&](size_t bytes) -> char* {
    char* p = ws + off;
    off = (off + bytes + 255) & ~(size_t)255;
    return p;
  };
  float* deg   = (float*)carve((size_t)N * 4);
  float* xacc  = (float*)carve((size_t)N * 12 * 4);
  float* agg   = (float*)carve((size_t)N * HIDX * 4);
  float* hb[2] = { (float*)carve((size_t)N * DF * 4), (float*)carve((size_t)N * DF * 4) };
  float* xb[2] = { (float*)carve((size_t)N * 12 * 4), (float*)carve((size_t)N * 12 * 4) };

  struct LW { unsigned short *rad, *e1, *e2, *n1, *n2, *c1, *c2; } lw[3];
  for (int l = 0; l < 3; ++l) {
    lw[l].rad = (unsigned short*)carve((size_t)256 * 16 * 2);
    lw[l].e1  = (unsigned short*)carve((size_t)CIN1 * 256 * 2);
    lw[l].e2  = (unsigned short*)carve((size_t)256 * 256 * 2);
    lw[l].n1  = (unsigned short*)carve((size_t)CINN * 256 * 2);
    lw[l].n2  = (unsigned short*)carve((size_t)256 * 256 * 2);
    lw[l].c1  = (unsigned short*)carve((size_t)256 * 256 * 2);
    lw[l].c2  = (unsigned short*)carve((size_t)256 * 16 * 2);
  }

  // weight prep (every call; no cached state)
  for (int l = 0; l < 3; ++l) {
    int b = 7 + 15 * l;
    auto T = [&](int idx, unsigned short* dst, int K, int Nn) {
      int total = K * Nn;
      transpose_bf16<<<(total + 255) / 256, 256, 0, stream>>>((const float*)d_in[idx], dst, K, Nn);
    };
    T(b + 14, lw[l].rad, 256, 16);
    T(b + 10, lw[l].e1,  CIN1, 256);
    T(b + 11, lw[l].e2,  256, 256);
    T(b + 12, lw[l].n1,  CINN, 256);
    T(b + 13, lw[l].n2,  256, 256);
    T(b + 8,  lw[l].c1,  256, 256);
    T(b + 9,  lw[l].c2,  256, 16);
  }

  zero_f32<<<256, 256, 0, stream>>>(deg, (size_t)N);
  degree_kernel<<<(E + 255) / 256, 256, 0, stream>>>(el, deg, E);

  float* h_fin = (float*)d_out;
  float* x_fin = (float*)d_out + (size_t)N * DF;

  for (int l = 0; l < 3; ++l) {
    int b = 7 + 15 * l;
    const float* h_in = (l == 0) ? h0 : hb[(l + 1) & 1];
    const float* x_in = (l == 0) ? x0 : xb[(l + 1) & 1];
    float* h_out = (l == 2) ? h_fin : hb[l & 1];
    float* x_out = (l == 2) ? x_fin : xb[l & 1];

    zero_f32<<<512, 256, 0, stream>>>(agg,  (size_t)N * HIDX);
    zero_f32<<<256, 256, 0, stream>>>(xacc, (size_t)N * 12);

    dymean_edge_kernel<<<(E + 15) / 16, 128, 0, stream>>>(
        h_in, x_in, el, cattr, cw, ew, eattr,
        lw[l].rad, (const float*)d_in[b + 5],
        lw[l].e1,  (const float*)d_in[b + 1],
        lw[l].e2,  (const float*)d_in[b + 2],
        lw[l].c1,  (const float*)d_in[b + 0],
        lw[l].c2,
        agg, xacc, E);

    dymean_node_kernel<<<(N + 15) / 16, 128, 0, stream>>>(
        h_in, x_in, agg, xacc, deg,
        lw[l].n1, (const float*)d_in[b + 3],
        lw[l].n2, (const float*)d_in[b + 4],
        (const float*)d_in[b + 7], (const float*)d_in[b + 6],
        h_out, x_out, N);
  }
}
